// Decoder_70205535421004
// MI455X (gfx1250) — matrix operand order repaired and checked by reading, not compiled, not run on hardware
//
#include <hip/hip_runtime.h>

// ---------------- problem constants ----------------
#define B_    32
#define T_    512
#define L_    128
#define E_    1024
#define D_    1024
#define A_    512
#define V_    5000
#define C_    10
#define KF_   201
#define NSTEP 129          // L+1
#define VPAD  5120         // 40 * 128
#define MOUT  (NSTEP * B_) // 4128

typedef __attribute__((ext_vector_type(16))) __bf16 v16bf;
typedef __attribute__((ext_vector_type(8)))  float  v8f;

// ---------------- bf16 helpers (bit-exact RNE) ----------------
__device__ __forceinline__ __bf16 f2bf(float x) {
  union { float f; unsigned u; } v; v.f = x;
  unsigned r = v.u + 0x7FFFu + ((v.u >> 16) & 1u);
  unsigned short h = (unsigned short)(r >> 16);
  return __builtin_bit_cast(__bf16, h);
}
__device__ __forceinline__ float bf2f(__bf16 x) {
  unsigned short h = __builtin_bit_cast(unsigned short, x);
  union { unsigned u; float f; } v; v.u = ((unsigned)h) << 16;
  return v.f;
}
__device__ __forceinline__ float sigf(float x) { return 1.0f / (1.0f + expf(-x)); }

// ---------------- WMMA fragment loaders (wave32, 16x16x32 bf16) ----------------
// A fragment: 16x32 (MxK) row-major source. lanes 0-15: M=lane, elems 0..7->K+0..7,
// elems 8..15->K+16..23 ; lanes 16-31: K half offset +8 / +24.
__device__ __forceinline__ v16bf load_a_frag(const __bf16* __restrict__ A, int lda,
                                             int m0, int k0, int lane) {
  int m  = m0 + (lane & 15);
  int kh = (lane >> 4) * 8;
  const __bf16* p = A + (size_t)m * lda + k0 + kh;
  v16bf f;
#pragma unroll
  for (int e = 0; e < 8; ++e) { f[e] = p[e]; f[e + 8] = p[e + 16]; }
  return f;
}
__device__ __forceinline__ v16bf load_a_frag_clamp(const __bf16* __restrict__ A, int lda,
                                                   int m0, int k0, int lane, int mmax) {
  int m = m0 + (lane & 15); if (m > mmax) m = mmax;
  int kh = (lane >> 4) * 8;
  const __bf16* p = A + (size_t)m * lda + k0 + kh;
  v16bf f;
#pragma unroll
  for (int e = 0; e < 8; ++e) { f[e] = p[e]; f[e + 8] = p[e + 16]; }
  return f;
}
// B fragment from W stored [N,K] row-major (computing X @ W^T).
// lanes 0-15: N=lane, elems 0..7 -> K+0..7, elems 8..15 -> K+16..23 ; lanes 16-31: +8 / +24.
__device__ __forceinline__ v16bf load_b_frag(const __bf16* __restrict__ W, int ldw,
                                             int n0, int k0, int lane) {
  int n  = n0 + (lane & 15);
  int ks = (lane >> 4) * 8;
  const __bf16* p = W + (size_t)n * ldw + k0 + ks;
  v16bf f;
#pragma unroll
  for (int e = 0; e < 8; ++e) { f[e] = p[e]; f[e + 8] = p[e + 16]; }
  return f;
}
__device__ __forceinline__ v8f wmma_bf16(v16bf a, v16bf b, v8f c) {
  return __builtin_amdgcn_wmma_f32_16x16x32_bf16(false, a, false, b, (short)0, c, false, false);
}
// gfx1250 prefetch (lowers to global_prefetch_b8)
__device__ __forceinline__ void pf(const void* p) { __builtin_prefetch(p, 0, 3); }

// ---------------- converters ----------------
__global__ void cvt_bf16(const float* __restrict__ in, __bf16* __restrict__ out, int n) {
  int i = blockIdx.x * blockDim.x + threadIdx.x;
  if (i < n) out[i] = f2bf(in[i]);
}
// out_W [V,1024] -> padded bf16 [VPAD,1024] (zero rows >= V)
__global__ void cvt_outW(const float* __restrict__ W, __bf16* __restrict__ out) {
  int i = blockIdx.x * blockDim.x + threadIdx.x;
  if (i >= VPAD * D_) return;
  int r = i >> 10;
  out[i] = (r < V_) ? f2bf(W[i]) : f2bf(0.0f);
}

// ---------------- big GEMM: C[M,N] = A[M,K] @ W[N,K]^T + bias[n] ----------------
// block = 256 thr (8 waves); block tile 64(M) x 128(N); wave tile 32x32 (2x2 WMMA).
// M%64==0, N%128==0, K%32==0 assumed.
__global__ void gemm_bf16_64x128(const __bf16* __restrict__ A, const __bf16* __restrict__ W,
                                 const float* __restrict__ bias, float* __restrict__ C,
                                 int M, int N, int K) {
  int wave = threadIdx.x >> 5, lane = threadIdx.x & 31;
  int m0 = blockIdx.y * 64  + (wave & 1) * 32;
  int n0 = blockIdx.x * 128 + (wave >> 1) * 32;
  v8f acc[2][2] = {};
#pragma unroll 2
  for (int k = 0; k < K; k += 32) {
    if (k + 128 < K) {
      pf(A + (size_t)(m0 + (lane & 15)) * K + k + 128);
      pf(W + (size_t)(n0 + (lane & 15)) * K + k + 128);
    }
    v16bf a0 = load_a_frag(A, K, m0,      k, lane);
    v16bf a1 = load_a_frag(A, K, m0 + 16, k, lane);
    v16bf b0 = load_b_frag(W, K, n0,      k, lane);
    v16bf b1 = load_b_frag(W, K, n0 + 16, k, lane);
    acc[0][0] = wmma_bf16(a0, b0, acc[0][0]);
    acc[0][1] = wmma_bf16(a0, b1, acc[0][1]);
    acc[1][0] = wmma_bf16(a1, b0, acc[1][0]);
    acc[1][1] = wmma_bf16(a1, b1, acc[1][1]);
  }
  int nlo = lane & 15, mhi = (lane >> 4) * 8;
#pragma unroll
  for (int i = 0; i < 2; ++i)
#pragma unroll
    for (int j = 0; j < 2; ++j) {
      int n = n0 + j * 16 + nlo;
      float bv = bias ? bias[n] : 0.0f;
#pragma unroll
      for (int e = 0; e < 8; ++e) {
        int m = m0 + i * 16 + mhi + e;
        C[(size_t)m * N + n] = acc[i][j][e] + bv;
      }
    }
}

// ---------------- small-M dual GEMM (M=32): gates = A0@W0^T + A1@W1^T + b0 + b1 ----
// one wave per 32-wide N strip (2x2 WMMA); 1.0 frag-loads per WMMA.
__global__ void gemm_m32_dual(const __bf16* __restrict__ A0, const __bf16* __restrict__ W0, int K0,
                              const __bf16* __restrict__ A1, const __bf16* __restrict__ W1, int K1,
                              const float* __restrict__ b0, const float* __restrict__ b1,
                              float* __restrict__ C, int N) {
  int wave = threadIdx.x >> 5, lane = threadIdx.x & 31;
  int n0 = (blockIdx.x * 8 + wave) * 32;
  v8f acc[2][2] = {};   // [mTile][nStrip]
#pragma unroll 2
  for (int k = 0; k < K0; k += 32) {
    if (k + 128 < K0) {
      pf(W0 + (size_t)(n0 + (lane & 15)) * K0 + k + 128);
      pf(W0 + (size_t)(n0 + 16 + (lane & 15)) * K0 + k + 128);
    }
    v16bf aL = load_a_frag(A0, K0, 0,  k, lane);
    v16bf aH = load_a_frag(A0, K0, 16, k, lane);
    v16bf bA = load_b_frag(W0, K0, n0,      k, lane);
    v16bf bB = load_b_frag(W0, K0, n0 + 16, k, lane);
    acc[0][0] = wmma_bf16(aL, bA, acc[0][0]);
    acc[0][1] = wmma_bf16(aL, bB, acc[0][1]);
    acc[1][0] = wmma_bf16(aH, bA, acc[1][0]);
    acc[1][1] = wmma_bf16(aH, bB, acc[1][1]);
  }
  if (A1) {
#pragma unroll 2
    for (int k = 0; k < K1; k += 32) {
      if (k + 128 < K1) {
        pf(W1 + (size_t)(n0 + (lane & 15)) * K1 + k + 128);
        pf(W1 + (size_t)(n0 + 16 + (lane & 15)) * K1 + k + 128);
      }
      v16bf aL = load_a_frag(A1, K1, 0,  k, lane);
      v16bf aH = load_a_frag(A1, K1, 16, k, lane);
      v16bf bA = load_b_frag(W1, K1, n0,      k, lane);
      v16bf bB = load_b_frag(W1, K1, n0 + 16, k, lane);
      acc[0][0] = wmma_bf16(aL, bA, acc[0][0]);
      acc[0][1] = wmma_bf16(aL, bB, acc[0][1]);
      acc[1][0] = wmma_bf16(aH, bA, acc[1][0]);
      acc[1][1] = wmma_bf16(aH, bB, acc[1][1]);
    }
  }
  int nlo = lane & 15, mhi = (lane >> 4) * 8;
#pragma unroll
  for (int j = 0; j < 2; ++j) {
    int n = n0 + j * 16 + nlo;
    float bv = (b0 ? b0[n] : 0.0f) + (b1 ? b1[n] : 0.0f);
#pragma unroll
    for (int e = 0; e < 8; ++e) {
      C[(size_t)(mhi + e) * N + n]      = acc[0][j][e] + bv;
      C[(size_t)(16 + mhi + e) * N + n] = acc[1][j][e] + bv;
    }
  }
}

// ---------------- output GEMM: y[b,l,v] = zs[l*32+b,:] @ out_W^T + out_b ----------
__global__ void out_gemm(const __bf16* __restrict__ zs, const __bf16* __restrict__ Wp,
                         const float* __restrict__ ob, float* __restrict__ out) {
  int wave = threadIdx.x >> 5, lane = threadIdx.x & 31;
  int m0 = blockIdx.y * 64  + (wave & 1) * 32;
  int n0 = blockIdx.x * 128 + (wave >> 1) * 32;
  v8f acc[2][2] = {};
#pragma unroll 2
  for (int k = 0; k < D_; k += 32) {
    if (k + 128 < D_) {
      pf(Wp + (size_t)(n0 + (lane & 15)) * D_ + k + 128);
      pf(Wp + (size_t)(n0 + 16 + (lane & 15)) * D_ + k + 128);
    }
    v16bf a0 = load_a_frag_clamp(zs, D_, m0,      k, lane, MOUT - 1);
    v16bf a1 = load_a_frag_clamp(zs, D_, m0 + 16, k, lane, MOUT - 1);
    v16bf b0 = load_b_frag(Wp, D_, n0,      k, lane);
    v16bf b1 = load_b_frag(Wp, D_, n0 + 16, k, lane);
    acc[0][0] = wmma_bf16(a0, b0, acc[0][0]);
    acc[0][1] = wmma_bf16(a0, b1, acc[0][1]);
    acc[1][0] = wmma_bf16(a1, b0, acc[1][0]);
    acc[1][1] = wmma_bf16(a1, b1, acc[1][1]);
  }
  int nlo = lane & 15, mhi = (lane >> 4) * 8;
#pragma unroll
  for (int i = 0; i < 2; ++i)
#pragma unroll
    for (int j = 0; j < 2; ++j) {
      int v = n0 + j * 16 + nlo;
#pragma unroll
      for (int e = 0; e < 8; ++e) {
        int m = m0 + i * 16 + mhi + e;
        if (m < MOUT && v < V_) {
          int bb = m & 31, l = m >> 5;
          out[(size_t)bb * (NSTEP * V_) + (size_t)l * V_ + v] = acc[i][j][e] + ob[v];
        }
      }
    }
}

// ---------------- init ----------------
__global__ void init_aw(const int* __restrict__ hlens, float* __restrict__ aw) {
  int i = blockIdx.x * blockDim.x + threadIdx.x;
  if (i >= B_ * T_) return;
  int b = i >> 9, t = i & 511;
  aw[i] = (t < hlens[b]) ? 1.0f / (float)hlens[b] : 0.0f;
}
__global__ void init_state(float* z0, float* c0, float* z1, float* c1,
                           __bf16* z0b, __bf16* z1b) {
  int i = blockIdx.x * blockDim.x + threadIdx.x;
  if (i >= B_ * D_) return;
  z0[i] = 0.f; c0[i] = 0.f; z1[i] = 0.f; c1[i] = 0.f;
  z0b[i] = f2bf(0.f); z1b[i] = f2bf(0.f);
}

// ---------------- per-step kernels ----------------
__global__ void embed_gather(const int* __restrict__ ys, const float* __restrict__ embW,
                             __bf16* __restrict__ ey_bf, int step) {
  int i = blockIdx.x * blockDim.x + threadIdx.x;
  if (i >= B_ * D_) return;
  int b = i >> 10, d = i & 1023;
  int tok = (step == 0) ? (V_ - 1) : ys[b * L_ + step - 1];
  ey_bf[b * 2048 + d] = f2bf(embW[(size_t)tok * D_ + d]);
}

// conv[b,c,t] = sum_k aw[b, t+k-100] * convW[c,k]
__global__ void conv_aw(const float* __restrict__ aw, const float* __restrict__ convW,
                        float* __restrict__ conv) {
  int i = blockIdx.x * blockDim.x + threadIdx.x;
  if (i >= B_ * C_ * T_) return;
  int t = i & 511, c = (i >> 9) % C_, b = i / (C_ * T_);
  float s = 0.f;
  for (int k = 0; k < KF_; ++k) {
    int tt = t + k - (KF_ / 2);
    if (tt >= 0 && tt < T_) s += aw[b * T_ + tt] * convW[c * KF_ + k];
  }
  conv[i] = s;
}

// one wave per (b,t): e = sum_a tanh(pre_enc + conv_proj + dec_proj) * gvec
__global__ void attn_score(const float* __restrict__ pre_enc, const float* __restrict__ dec_proj,
                           const float* __restrict__ conv, const float* __restrict__ attW,
                           const float* __restrict__ gvecW, const float* __restrict__ gvecb,
                           const int* __restrict__ hlens, float* __restrict__ e_out) {
  int gwave = blockIdx.x * 8 + (threadIdx.x >> 5);
  int lane = threadIdx.x & 31;
  int b = gwave >> 9, t = gwave & 511;
  float cv[C_];
#pragma unroll
  for (int c = 0; c < C_; ++c) cv[c] = conv[(b * C_ + c) * T_ + t];
  const float* pe = pre_enc + (size_t)(b * T_ + t) * A_;
  const float* dp = dec_proj + b * A_;
  float part = 0.f;
#pragma unroll
  for (int i = 0; i < A_ / 32; ++i) {
    int a = i * 32 + lane;
    float val = pe[a] + dp[a];
#pragma unroll
    for (int c = 0; c < C_; ++c) val += cv[c] * attW[a * C_ + c];
    part += tanhf(val) * gvecW[a];
  }
#pragma unroll
  for (int off = 16; off > 0; off >>= 1) part += __shfl_xor(part, off, 32);
  if (lane == 0)
    e_out[b * T_ + t] = (t < hlens[b]) ? (part + gvecb[0]) : -1e9f;
}

// softmax over T with SCALING=2 ; one block (256) per batch row
__global__ void softmax_aw(const float* __restrict__ e, float* __restrict__ aw) {
  int b = blockIdx.x, tid = threadIdx.x;
  __shared__ float rmax[8], rsum[8];
  float m = -1e30f;
  for (int t = tid; t < T_; t += 256) m = fmaxf(m, e[b * T_ + t]);
#pragma unroll
  for (int off = 16; off > 0; off >>= 1) m = fmaxf(m, __shfl_xor(m, off, 32));
  if ((tid & 31) == 0) rmax[tid >> 5] = m;
  __syncthreads();
  m = rmax[0];
#pragma unroll
  for (int i = 1; i < 8; ++i) m = fmaxf(m, rmax[i]);
  float s = 0.f;
  for (int t = tid; t < T_; t += 256) s += expf(2.0f * (e[b * T_ + t] - m));
#pragma unroll
  for (int off = 16; off > 0; off >>= 1) s += __shfl_xor(s, off, 32);
  if ((tid & 31) == 0) rsum[tid >> 5] = s;
  __syncthreads();
  s = 0.f;
#pragma unroll
  for (int i = 0; i < 8; ++i) s += rsum[i];
  float inv = 1.0f / s;
  for (int t = tid; t < T_; t += 256)
    aw[b * T_ + t] = expf(2.0f * (e[b * T_ + t] - m)) * inv;
}

// att_c[b,e] = sum_t aw[b,t] * hs[b,t,e] ; written as bf16 into ey[:,1024:2048]
__global__ void att_context(const float* __restrict__ aw, const __bf16* __restrict__ hs_bf,
                            __bf16* __restrict__ ey_bf) {
  int i = blockIdx.x * blockDim.x + threadIdx.x;
  if (i >= B_ * E_) return;
  int b = i >> 10, d = i & 1023;
  const __bf16* h = hs_bf + (size_t)b * T_ * E_ + d;
  float s = 0.f;
  for (int t = 0; t < T_; ++t) s += aw[b * T_ + t] * bf2f(h[(size_t)t * E_]);
  ey_bf[b * 2048 + E_ + d] = f2bf(s);
}

// LSTM elementwise: gates [32,4096] (i,f,g,o) -> c,h ; also bf16 copy (+ optional zs slot)
__global__ void lstm_update(const float* __restrict__ gates, float* __restrict__ c,
                            float* __restrict__ z, __bf16* __restrict__ z_bf,
                            __bf16* __restrict__ zs_slot) {
  int i = blockIdx.x * blockDim.x + threadIdx.x;
  if (i >= B_ * D_) return;
  int b = i >> 10, d = i & 1023;
  const float* g = gates + (size_t)b * 4 * D_;
  float ig = sigf(g[d]);
  float fg = sigf(g[d + D_]);
  float gg = tanhf(g[d + 2 * D_]);
  float og = sigf(g[d + 3 * D_]);
  float cn = fg * c[i] + ig * gg;
  float h = og * tanhf(cn);
  c[i] = cn; z[i] = h;
  __bf16 hb = f2bf(h);
  z_bf[i] = hb;
  if (zs_slot) zs_slot[i] = hb;
}

// ---------------- host launcher ----------------
extern "C" void kernel_launch(void* const* d_in, const int* in_sizes, int n_in,
                              void* d_out, int out_size, void* d_ws, size_t ws_size,
                              hipStream_t stream) {
  const float* hs       = (const float*)d_in[0];
  const int*   hlens    = (const int*)  d_in[1];
  const int*   ys       = (const int*)  d_in[2];
  const float* embW     = (const float*)d_in[3];
  const float* w_ih0    = (const float*)d_in[4];
  const float* w_hh0    = (const float*)d_in[5];
  const float* b_ih0    = (const float*)d_in[6];
  const float* b_hh0    = (const float*)d_in[7];
  const float* w_ih1    = (const float*)d_in[8];
  const float* w_hh1    = (const float*)d_in[9];
  const float* b_ih1    = (const float*)d_in[10];
  const float* b_hh1    = (const float*)d_in[11];
  const float* attEncW  = (const float*)d_in[12];
  const float* attEncB  = (const float*)d_in[13];
  const float* attDecW  = (const float*)d_in[14];
  const float* attConvW = (const float*)d_in[15];
  const float* attAttW  = (const float*)d_in[16];
  const float* gvecW    = (const float*)d_in[17];
  const float* gvecB    = (const float*)d_in[18];
  const float* outW     = (const float*)d_in[19];
  const float* outB     = (const float*)d_in[20];
  float* out = (float*)d_out;

  char* ws = (char*)d_ws;
  size_t off = 0;
  auto take = [&](size_t bytes) -> void* {
    void* p = ws + off; off += (bytes + 255) & ~(size_t)255; return p;
  };
  __bf16* hs_bf    = (__bf16*)take((size_t)B_ * T_ * E_ * 2);
  __bf16* encW_bf  = (__bf16*)take((size_t)A_ * E_ * 2);
  __bf16* decW_bf  = (__bf16*)take((size_t)A_ * D_ * 2);
  __bf16* wih0_bf  = (__bf16*)take((size_t)4 * D_ * 2048 * 2);
  __bf16* whh0_bf  = (__bf16*)take((size_t)4 * D_ * D_ * 2);
  __bf16* wih1_bf  = (__bf16*)take((size_t)4 * D_ * D_ * 2);
  __bf16* whh1_bf  = (__bf16*)take((size_t)4 * D_ * D_ * 2);
  __bf16* outW_bf  = (__bf16*)take((size_t)VPAD * D_ * 2);
  float*  pre_enc  = (float*) take((size_t)B_ * T_ * A_ * 4);
  float*  aw       = (float*) take((size_t)B_ * T_ * 4);
  float*  e_buf    = (float*) take((size_t)B_ * T_ * 4);
  float*  conv     = (float*) take((size_t)B_ * C_ * T_ * 4);
  float*  dec_proj = (float*) take((size_t)B_ * A_ * 4);
  __bf16* ey_bf    = (__bf16*)take((size_t)B_ * 2048 * 2);
  float*  gates    = (float*) take((size_t)B_ * 4 * D_ * 4);
  float*  z0 = (float*)take((size_t)B_ * D_ * 4);
  float*  c0 = (float*)take((size_t)B_ * D_ * 4);
  float*  z1 = (float*)take((size_t)B_ * D_ * 4);
  float*  c1 = (float*)take((size_t)B_ * D_ * 4);
  __bf16* z0_bf = (__bf16*)take((size_t)B_ * D_ * 2);
  __bf16* z1_bf = (__bf16*)take((size_t)B_ * D_ * 2);
  __bf16* zs_bf = (__bf16*)take((size_t)NSTEP * B_ * D_ * 2);

  dim3 blk(256);
  // ---- one-time conversions (deterministic each call) ----
  cvt_bf16<<<(B_*T_*E_ + 255) / 256, blk, 0, stream>>>(hs, hs_bf, B_*T_*E_);
  cvt_bf16<<<(A_*E_ + 255) / 256, blk, 0, stream>>>(attEncW, encW_bf, A_*E_);
  cvt_bf16<<<(A_*D_ + 255) / 256, blk, 0, stream>>>(attDecW, decW_bf, A_*D_);
  cvt_bf16<<<(4*D_*2048 + 255) / 256, blk, 0, stream>>>(w_ih0, wih0_bf, 4*D_*2048);
  cvt_bf16<<<(4*D_*D_ + 255) / 256, blk, 0, stream>>>(w_hh0, whh0_bf, 4*D_*D_);
  cvt_bf16<<<(4*D_*D_ + 255) / 256, blk, 0, stream>>>(w_ih1, wih1_bf, 4*D_*D_);
  cvt_bf16<<<(4*D_*D_ + 255) / 256, blk, 0, stream>>>(w_hh1, whh1_bf, 4*D_*D_);
  cvt_outW<<<(VPAD*D_ + 255) / 256, blk, 0, stream>>>(outW, outW_bf);

  // ---- pre_enc = hs @ att_enc_W^T + b : M=16384 N=512 K=1024 ----
  gemm_bf16_64x128<<<dim3(A_/128, (B_*T_)/64), blk, 0, stream>>>(
      hs_bf, encW_bf, attEncB, pre_enc, B_*T_, A_, E_);

  init_aw<<<(B_*T_ + 255) / 256, blk, 0, stream>>>(hlens, aw);
  init_state<<<(B_*D_ + 255) / 256, blk, 0, stream>>>(z0, c0, z1, c1, z0_bf, z1_bf);

  // ---- recurrent loop ----
  for (int step = 0; step < NSTEP; ++step) {
    embed_gather<<<(B_*D_ + 255) / 256, blk, 0, stream>>>(ys, embW, ey_bf, step);
    // dec_proj = z0 @ att_dec_W^T  (N=512, 256 cols per block)
    gemm_m32_dual<<<A_/256, blk, 0, stream>>>(z0_bf, decW_bf, D_,
                                              nullptr, nullptr, 0,
                                              nullptr, nullptr, dec_proj, A_);
    conv_aw<<<(B_*C_*T_ + 255) / 256, blk, 0, stream>>>(aw, attConvW, conv);
    attn_score<<<(B_*T_) / 8, blk, 0, stream>>>(pre_enc, dec_proj, conv, attAttW,
                                                gvecW, gvecB, hlens, e_buf);
    softmax_aw<<<B_, blk, 0, stream>>>(e_buf, aw);
    att_context<<<(B_*E_ + 255) / 256, blk, 0, stream>>>(aw, hs_bf, ey_bf);
    // LSTM0: gates = ey @ w_ih0^T + z0 @ w_hh0^T + biases  (N=4096)
    gemm_m32_dual<<<(4*D_)/256, blk, 0, stream>>>(ey_bf, wih0_bf, 2048,
                                                  z0_bf, whh0_bf, D_,
                                                  b_ih0, b_hh0, gates, 4*D_);
    lstm_update<<<(B_*D_ + 255) / 256, blk, 0, stream>>>(gates, c0, z0, z0_bf, nullptr);
    // LSTM1
    gemm_m32_dual<<<(4*D_)/256, blk, 0, stream>>>(z0_bf, wih1_bf, D_,
                                                  z1_bf, whh1_bf, D_,
                                                  b_ih1, b_hh1, gates, 4*D_);
    lstm_update<<<(B_*D_ + 255) / 256, blk, 0, stream>>>(gates, c1, z1, z1_bf,
                                                         zs_bf + (size_t)step * B_ * D_);
  }

  // ---- y[b,l,v] = zs @ out_W^T + out_b : M=4128 Npad=5120 K=1024 ----
  out_gemm<<<dim3(VPAD/128, (MOUT + 63) / 64), blk, 0, stream>>>(zs_bf, outW_bf, outB, out);
}
